// MyLSTM_53386443489404
// MI455X (gfx1250) — matrix operand order repaired and checked by reading, not compiled, not run on hardware
//
#include <hip/hip_runtime.h>
#include <hip/hip_bf16.h>
#include <cstdint>
#include <cstddef>

// Problem dims (fixed by reference)
#define B_   64
#define T_   1024
#define V_   512
#define H_   512
#define NH4  2048                      // 4*H (f,i,o,c packed)
#define BH   (B_ * H_)                 // 32768
#define BTV  ((size_t)B_ * T_ * V_)    // 33,554,432

// ---------------- WMMA types / helpers (CDNA5 wave32) ----------------
typedef __attribute__((ext_vector_type(16))) __bf16 v16bf;
typedef __attribute__((ext_vector_type(8)))  float  v8f;
typedef __attribute__((ext_vector_type(4)))  unsigned int u32x4;

union FragU { v16bf v; u32x4 q[2]; };

__device__ inline v8f zero8() {
  v8f z;
#pragma unroll
  for (int i = 0; i < 8; ++i) z[i] = 0.f;
  return z;
}

// round-to-nearest-even fp32 -> bf16 bits
__device__ inline unsigned short f2bf(float f) {
  union { float f; unsigned int u; } c; c.f = f;
  unsigned int u = c.u + 0x7FFFu + ((c.u >> 16) & 1u);
  return (unsigned short)(u >> 16);
}

// A-matrix 16x32 bf16 frag per ISA layout:
// lane l: M = l&15; elements 0..7  -> k = kb + (l>>4)*8 .. +7
//                   elements 8..15 -> k = kb + 16 + (l>>4)*8 .. +7
__device__ inline v16bf loadA(const unsigned short* rowBaseKb, int hl) {
  FragU f;
  const u32x4* p = (const u32x4*)(rowBaseKb + hl * 8);
  f.q[0] = p[0];      // 8 bf16
  f.q[1] = p[2];      // +16 elements = +32 bytes = +2 dwordx4
  return f.v;
}

// B-matrix 32x16 bf16 frag: lane l holds column n = l&15,
// elements 0..7 -> k = kb + (l>>4)*8 + i, 8..15 -> k = kb + 16 + (l>>4)*8 + (i-8)  (K-contiguous => weights stored transposed)
__device__ inline v16bf loadB(const unsigned short* nRowKb, int hl) {
  FragU f;
  const u32x4* p = (const u32x4*)(nRowKb + hl * 8);
  f.q[0] = p[0];
  f.q[1] = p[2];
  return f.v;
}

__device__ inline v8f wmma_bf16(v16bf a, v16bf b, v8f c) {
  // (neg_a, A, neg_b, B, c_mod, C, reuse_a, reuse_b)
  return __builtin_amdgcn_wmma_f32_16x16x32_bf16(false, a, false, b, (short)0, c,
                                                 false, false);
}

// ---------------- small prep kernels ----------------
__global__ __launch_bounds__(256) void convert_x(const float* __restrict__ X,
                                                 unsigned short* __restrict__ Xbf) {
  size_t i = ((size_t)blockIdx.x * 256 + threadIdx.x) * 4;
  float4 f = *(const float4*)(X + i);
  unsigned int lo = (unsigned)f2bf(f.x) | ((unsigned)f2bf(f.y) << 16);
  unsigned int hi = (unsigned)f2bf(f.z) | ((unsigned)f2bf(f.w) << 16);
  *(uint2*)(Xbf + i) = make_uint2(lo, hi);
}

// Pack W_x/W_h transposed: WT[n][k] = w[k][n_local]  (n in [0,2048): gate = n>>9)
// Also WhyT[v][k] = w_hy[k][v] and combined input bias biasx[2048].
__global__ __launch_bounds__(256) void pack_weights(
    const float* __restrict__ wxf, const float* __restrict__ wxi,
    const float* __restrict__ wxo, const float* __restrict__ wxc,
    const float* __restrict__ whf, const float* __restrict__ whi,
    const float* __restrict__ who, const float* __restrict__ whc,
    const float* __restrict__ why,
    const float* __restrict__ bf, const float* __restrict__ bi,
    const float* __restrict__ bo, const float* __restrict__ bc,
    unsigned short* __restrict__ WxT, unsigned short* __restrict__ WhT,
    unsigned short* __restrict__ WhyT, float* __restrict__ biasx) {
  const int idx = blockIdx.x * 256 + threadIdx.x;   // 0 .. 2048*512-1
  const int n  = idx >> 9;
  const int k  = idx & 511;
  const int g  = n >> 9;
  const int nl = n & 511;
  const float* wx = (g == 0) ? wxf : (g == 1) ? wxi : (g == 2) ? wxo : wxc;
  const float* wh = (g == 0) ? whf : (g == 1) ? whi : (g == 2) ? who : whc;
  WxT[(size_t)n * H_ + k] = f2bf(wx[(size_t)k * H_ + nl]);
  WhT[(size_t)n * H_ + k] = f2bf(wh[(size_t)k * H_ + nl]);
  if (n < V_) WhyT[(size_t)n * H_ + k] = f2bf(why[(size_t)k * V_ + n]);
  if (k == 0) {
    const float* bb = (g == 0) ? bf : (g == 1) ? bi : (g == 2) ? bo : bc;
    biasx[n] = bb[nl];
  }
}

__global__ __launch_bounds__(256) void init_state(
    const float* __restrict__ st, const float* __restrict__ cand,
    unsigned short* __restrict__ hbf, float* __restrict__ hF,
    float* __restrict__ cF, int* __restrict__ cnt) {
  int i = blockIdx.x * 256 + threadIdx.x;   // 0..BH-1
  hbf[i] = f2bf(st[i]);
  hF[i]  = st[i];
  cF[i]  = cand[i];
  if (i == 0) *cnt = 0;                     // reset grid-barrier counter each launch
}

// ---------------- phase 1: gate preactivations G[t][b][0:2048] ----------------
// C = Xbf[BT,512] @ WxT^T + biasx ; each wave: one 16-row M tile x four 16-col N
// tiles. Software-pipelined: frags for step k+1 are loaded before the WMMAs of k.
__global__ __launch_bounds__(256) void lstm_xproj(
    const unsigned short* __restrict__ Xbf, const unsigned short* __restrict__ WxT,
    const float* __restrict__ biasx, float* __restrict__ G) {
  const int lane = threadIdx.x & 31;
  const int hl   = lane >> 4;
  const int l15  = lane & 15;
  const size_t wv = (size_t)blockIdx.x * 8 + (threadIdx.x >> 5);  // 0..131071
  const int mt = (int)(wv >> 5);   // 0..4095 : 16 rows of (b*T + t)
  const int ng = (int)(wv & 31);   // 0..31   : 64 output columns
  const unsigned short* aRow = Xbf + (size_t)((mt << 4) + l15) * V_;
  const unsigned short* b0r = WxT + (size_t)((ng << 6) + l15) * H_;
  const unsigned short* b1r = b0r + 16 * H_;
  const unsigned short* b2r = b0r + 32 * H_;
  const unsigned short* b3r = b0r + 48 * H_;
  v8f a0 = zero8(), a1 = zero8(), a2 = zero8(), a3 = zero8();

  v16bf a  = loadA(aRow, hl);
  v16bf f0 = loadB(b0r, hl), f1 = loadB(b1r, hl);
  v16bf f2 = loadB(b2r, hl), f3 = loadB(b3r, hl);
  for (int kb = 0; kb < V_ - 32; kb += 32) {
    v16bf an = loadA(aRow + kb + 32, hl);
    v16bf n0 = loadB(b0r + kb + 32, hl), n1 = loadB(b1r + kb + 32, hl);
    v16bf n2 = loadB(b2r + kb + 32, hl), n3 = loadB(b3r + kb + 32, hl);
    a0 = wmma_bf16(a, f0, a0);
    a1 = wmma_bf16(a, f1, a1);
    a2 = wmma_bf16(a, f2, a2);
    a3 = wmma_bf16(a, f3, a3);
    a = an; f0 = n0; f1 = n1; f2 = n2; f3 = n3;
  }
  a0 = wmma_bf16(a, f0, a0);
  a1 = wmma_bf16(a, f1, a1);
  a2 = wmma_bf16(a, f2, a2);
  a3 = wmma_bf16(a, f3, a3);

#pragma unroll
  for (int tix = 0; tix < 4; ++tix) {
    v8f acc = (tix == 0) ? a0 : (tix == 1) ? a1 : (tix == 2) ? a2 : a3;
    const int n = (ng << 6) + (tix << 4) + l15;
    const float bias = biasx[n];
#pragma unroll
    for (int v = 0; v < 8; ++v) {
      const int rr = (mt << 4) + (hl << 3) + v;   // row = b*T + t
      const int bb = rr >> 10;
      const int tt = rr & (T_ - 1);
      G[(((size_t)tt * B_) + bb) * NH4 + n] = acc[v] + bias;
    }
  }
}

// ---------------- phase 2: persistent recurrent scan ----------------
#define NWG_SCAN 32   // 32 WGs * 8 waves = 256 waves = 128 gate units + 128 y units

__device__ inline void grid_sync(int* cnt, int target) {
  __threadfence();
  __syncthreads();
  if (threadIdx.x == 0) {
    __hip_atomic_fetch_add(cnt, 1, __ATOMIC_RELEASE, __HIP_MEMORY_SCOPE_AGENT);
    while (__hip_atomic_load(cnt, __ATOMIC_ACQUIRE, __HIP_MEMORY_SCOPE_AGENT) < target) {
      __builtin_amdgcn_s_sleep(8);
    }
  }
  __syncthreads();
  __threadfence();
}

// Iteration t (t = 0..T inclusive):
//   gate waves (0..127): read h_t (buf t&1), G[t], c_t -> write h_{t+1}, c_{t+1}   [skip t==T]
//   y    waves (128..255): y_{t-1} = h_t @ w_hy + b_y                              [skip t==0]
// One grid barrier per iteration. Epilogue operands are prefetched before the K
// loop so their latency hides under the WMMA chain.
__global__ __launch_bounds__(256) void lstm_scan(
    const unsigned short* __restrict__ WhT, const unsigned short* __restrict__ WhyT,
    const float* __restrict__ G, const float* __restrict__ b_y,
    unsigned short* hbf /* [2][B*H] bf16 ping-pong */,
    float* __restrict__ out /* [T*B, V] */, float* __restrict__ hF,
    float* __restrict__ cF, int* cnt) {
  const int lane = threadIdx.x & 31;
  const int hl   = lane >> 4;
  const int l15  = lane & 15;
  const int wv   = blockIdx.x * 8 + (threadIdx.x >> 5);   // 0..255

  for (int t = 0; t <= T_; ++t) {
    const unsigned short* hin = hbf + (size_t)(t & 1) * BH;
    if (wv < 128) {
      if (t < T_) {
        const int m = wv & 3;                 // batch tile (16 rows)
        const int j = wv >> 2;                // H column tile (16 cols)
        const unsigned short* aRow = hin + (size_t)((m << 4) + l15) * H_;
        const int ncol = (j << 4) + l15;
        const unsigned short* bFr = WhT + (size_t)(0 * H_ + ncol) * H_;
        const unsigned short* bIr = WhT + (size_t)(1 * H_ + ncol) * H_;
        const unsigned short* bOr = WhT + (size_t)(2 * H_ + ncol) * H_;
        const unsigned short* bCr = WhT + (size_t)(3 * H_ + ncol) * H_;

        // --- prefetch epilogue operands (consumed only after the GEMM) ---
        const float* g = G + (size_t)t * B_ * NH4;
        float gfv[8], giv[8], gov[8], gcv[8], cold[8];
#pragma unroll
        for (int v = 0; v < 8; ++v) {
          const int b = (m << 4) + (hl << 3) + v;
          const float* gb = g + (size_t)b * NH4;
          gfv[v]  = gb[ncol];
          giv[v]  = gb[H_ + ncol];
          gov[v]  = gb[2 * H_ + ncol];
          gcv[v]  = gb[3 * H_ + ncol];
          cold[v] = cF[b * H_ + ncol];
        }

        // --- pipelined K loop: 64 WMMAs, loads one step ahead ---
        v8f aF = zero8(), aI = zero8(), aO = zero8(), aC = zero8();
        v16bf a  = loadA(aRow, hl);
        v16bf fF = loadB(bFr, hl), fI = loadB(bIr, hl);
        v16bf fO = loadB(bOr, hl), fC = loadB(bCr, hl);
        for (int kb = 0; kb < H_ - 32; kb += 32) {
          v16bf an = loadA(aRow + kb + 32, hl);
          v16bf nF = loadB(bFr + kb + 32, hl), nI = loadB(bIr + kb + 32, hl);
          v16bf nO = loadB(bOr + kb + 32, hl), nC = loadB(bCr + kb + 32, hl);
          aF = wmma_bf16(a, fF, aF);
          aI = wmma_bf16(a, fI, aI);
          aO = wmma_bf16(a, fO, aO);
          aC = wmma_bf16(a, fC, aC);
          a = an; fF = nF; fI = nI; fO = nO; fC = nC;
        }
        aF = wmma_bf16(a, fF, aF);
        aI = wmma_bf16(a, fI, aI);
        aO = wmma_bf16(a, fO, aO);
        aC = wmma_bf16(a, fC, aC);

        unsigned short* hout = hbf + (size_t)((t + 1) & 1) * BH;
#pragma unroll
        for (int v = 0; v < 8; ++v) {
          const int b = (m << 4) + (hl << 3) + v;   // C layout: M = v + half*8
          float pf = aF[v] + gfv[v];
          float pi = aI[v] + giv[v];
          float po = aO[v] + gov[v];
          float pc = aC[v] + gcv[v];
          float fg = 1.f / (1.f + __expf(-pf));
          float ig = 1.f / (1.f + __expf(-pi));
          float og = 1.f / (1.f + __expf(-po));
          float ch = tanhf(pc);
          const int idx = b * H_ + ncol;
          float cn = fg * cold[v] + ig * ch;
          float hn = og * tanhf(cn);
          cF[idx]   = cn;
          hout[idx] = f2bf(hn);
          if (t == T_ - 1) hF[idx] = hn;   // fp32 h_final only on last step
        }
      }
    } else {
      if (t > 0) {
        const int u = wv - 128;
        const int m = u & 3;
        const int jv = u >> 2;                // V column tile
        const unsigned short* aRow = hin + (size_t)((m << 4) + l15) * H_;
        const int ncol = (jv << 4) + l15;
        const unsigned short* bYr = WhyT + (size_t)ncol * H_;
        const float by = b_y[ncol];           // prefetched before K loop

        v8f acc = zero8();
        v16bf a  = loadA(aRow, hl);
        v16bf fY = loadB(bYr, hl);
        for (int kb = 0; kb < H_ - 32; kb += 32) {
          v16bf an = loadA(aRow + kb + 32, hl);
          v16bf nY = loadB(bYr + kb + 32, hl);
          acc = wmma_bf16(a, fY, acc);
          a = an; fY = nY;
        }
        acc = wmma_bf16(a, fY, acc);

        float* yrow = out + (size_t)(t - 1) * B_ * V_;
#pragma unroll
        for (int v = 0; v < 8; ++v) {
          const int b = (m << 4) + (hl << 3) + v;
          yrow[(size_t)b * V_ + ncol] = acc[v] + by;
        }
      }
    }
    grid_sync(cnt, NWG_SCAN * (t + 1));
  }
}

// ---------------- host launch ----------------
extern "C" void kernel_launch(void* const* d_in, const int* in_sizes, int n_in,
                              void* d_out, int out_size, void* d_ws, size_t ws_size,
                              hipStream_t stream) {
  const float* X    = (const float*)d_in[0];
  const float* st   = (const float*)d_in[1];
  const float* cand = (const float*)d_in[2];
  const float* wxf  = (const float*)d_in[3];
  const float* whf  = (const float*)d_in[4];
  const float* bf   = (const float*)d_in[5];
  const float* wxi  = (const float*)d_in[6];
  const float* whi  = (const float*)d_in[7];
  const float* bi   = (const float*)d_in[8];
  const float* wxo  = (const float*)d_in[9];
  const float* who  = (const float*)d_in[10];
  const float* bo   = (const float*)d_in[11];
  const float* wxc  = (const float*)d_in[12];
  const float* whc  = (const float*)d_in[13];
  const float* bc   = (const float*)d_in[14];
  const float* why  = (const float*)d_in[15];
  const float* by   = (const float*)d_in[16];

  // Workspace layout (all 256B-aligned):
  //   [cnt][Xbf 64MB][WxT 2MB][WhT 2MB][WhyT 0.5MB][biasx 8KB][hbf 128KB][G 512MB]
  char* ws = (char*)d_ws;
  int* cnt = (int*)ws;
  unsigned short* Xbf  = (unsigned short*)(ws + 256);
  unsigned short* WxT  = (unsigned short*)(ws + 256 + BTV * 2);
  unsigned short* WhT  = WxT + (size_t)NH4 * H_;
  unsigned short* WhyT = WhT + (size_t)NH4 * H_;
  float* biasx         = (float*)(WhyT + (size_t)V_ * H_);
  unsigned short* hbf  = (unsigned short*)((char*)biasx + NH4 * sizeof(float));
  float* G             = (float*)((char*)hbf + 2 * BH * sizeof(unsigned short));

  float* out = (float*)d_out;       // [T*B, V]
  float* hF  = out + BTV;           // h_final [B,H]
  float* cF  = hF + BH;             // c_final [B,H] (doubles as live cell state)

  convert_x   <<<(int)(BTV / 4 / 256), 256, 0, stream>>>(X, Xbf);
  pack_weights<<<(NH4 * H_) / 256,     256, 0, stream>>>(wxf, wxi, wxo, wxc,
                                                         whf, whi, who, whc, why,
                                                         bf, bi, bo, bc,
                                                         WxT, WhT, WhyT, biasx);
  init_state  <<<BH / 256,             256, 0, stream>>>(st, cand, hbf, hF, cF, cnt);
  lstm_xproj  <<<16384,                256, 0, stream>>>(Xbf, WxT, biasx, G);
  lstm_scan   <<<NWG_SCAN,             256, 0, stream>>>(WhT, WhyT, G, by, hbf,
                                                         out, hF, cF, cnt);
}
